// MultiHeadAttention_377957122710
// MI455X (gfx1250) — matrix operand order repaired and checked by reading, not compiled, not run on hardware
//
#include <hip/hip_runtime.h>
#include <hip/hip_bf16.h>

// ---------------------------------------------------------------------------
// MHA forward for MI455X (gfx1250, wave32, WMMA bf16 16x16x32, f32 accum)
// B=4, T=2048, D=1024, H=16, HD=64
// GEMMs: 32x64 wave tiles, direct-global B-frags (L2-resident weights).
// Attention: double-buffered TDM (tensor_load_to_lds) stages K/V tiles into
// LDS overlapped with WMMA compute; TENSORcnt/DScnt fencing per CDNA5 ISA.
// ---------------------------------------------------------------------------

#define B_  4
#define T_  2048
#define D_  1024
#define H_  16
#define HD_ 64
#define M_  (B_ * T_)   // 8192

typedef __attribute__((ext_vector_type(16))) __bf16 v16bf;
typedef __attribute__((ext_vector_type(8)))  __bf16 v8bf;
typedef __attribute__((ext_vector_type(8)))  float  v8f;
typedef __attribute__((ext_vector_type(4)))  unsigned int v4u;
typedef __attribute__((ext_vector_type(8)))  int    v8i;
typedef __attribute__((ext_vector_type(4)))  int    v4i;

#if defined(__has_builtin)
#if __has_builtin(__builtin_amdgcn_tensor_load_to_lds)
#define HAVE_TDM 1
#endif
#endif

#if defined(__has_include)
#if __has_include(<hip/amd_detail/amd_gfx1250_TDM.h>)
#define TDM_6ARG 1
#endif
#endif

static __device__ __forceinline__ unsigned short f2bf(float f) {
    unsigned int u = __float_as_uint(f);
    u += 0x7fffu + ((u >> 16) & 1u);            // round-to-nearest-even
    return (unsigned short)(u >> 16);
}

static __device__ __forceinline__ v16bf cat8(v8bf lo, v8bf hi) {
    return __builtin_shufflevector(lo, hi, 0,1,2,3,4,5,6,7,8,9,10,11,12,13,14,15);
}

// A-fragment (16x32 bf16): two contiguous 8-elem runs at +0 and +8 elems
static __device__ __forceinline__ v16bf ld_afrag(const unsigned short* p) {
    v8bf lo = *(const v8bf*)(p);
    v8bf hi = *(const v8bf*)(p + 8);
    return cat8(lo, hi);
}

static __device__ __forceinline__ v8f wmma_bf16(v16bf a, v16bf b, v8f c) {
    return __builtin_amdgcn_wmma_f32_16x16x32_bf16(
        false, a, false, b, (short)0, c, false, false);
}

#ifdef HAVE_TDM
// --- Tensor DMA descriptor builders (ISA cdna5 §8.3/8.4) -------------------
// group0: [1:0]=count=1, [63:32]=lds byte addr, [120:64]=global byte addr,
//         [127:126]=type=2
static __device__ __forceinline__ v4u tdm_g0(unsigned lds_addr,
                                             unsigned long long gaddr) {
    v4u g0;
    g0[0] = 1u;
    g0[1] = lds_addr;
    g0[2] = (unsigned)gaddr;
    g0[3] = ((unsigned)(gaddr >> 32) & 0x01FFFFFFu) | (2u << 30);
    return g0;
}
// group1: data_size=3 (8B elems), mask=0, no pad/iterate.
// tensor_dim0[79:48], tensor_dim1[111:80], tile_dim0[127:112],
// tile_dim1[143:128], tile_dim2[159:144]=0, stride0[207:160], stride1=0.
static __device__ __forceinline__ v8i tdm_g1(unsigned td0, unsigned td1,
                                             unsigned tile0, unsigned tile1,
                                             unsigned stride0) {
    v8i g1;
    g1[0] = (int)(3u << 16);
    g1[1] = (int)((td0 & 0xFFFFu) << 16);
    g1[2] = (int)(((td0 >> 16) & 0xFFFFu) | ((td1 & 0xFFFFu) << 16));
    g1[3] = (int)(((td1 >> 16) & 0xFFFFu) | ((tile0 & 0xFFFFu) << 16));
    g1[4] = (int)(tile1 & 0xFFFFu);
    g1[5] = (int)stride0;
    g1[6] = 0;
    g1[7] = 0;
    return g1;
}

static __device__ __forceinline__ void tdm_load(v4u g0, v8i g1) {
    v4i z4 = {0, 0, 0, 0};
#ifdef TDM_6ARG
    v8i z8 = {0, 0, 0, 0, 0, 0, 0, 0};
    __builtin_amdgcn_tensor_load_to_lds(g0, g1, z4, z4, z8, 0);
#else
    __builtin_amdgcn_tensor_load_to_lds(g0, g1, z4, z4, 0);
#endif
}
#endif // HAVE_TDM

// ---------------------------------------------------------------------------
// 0) f32 -> bf16 conversion
// ---------------------------------------------------------------------------
__global__ void cvt_bf16_kernel(const float* __restrict__ src,
                                unsigned short* __restrict__ dst, int n) {
    int i = blockIdx.x * blockDim.x + threadIdx.x;
    if (i < n) dst[i] = f2bf(src[i]);
}

// ---------------------------------------------------------------------------
// 1) QKV projection GEMM:  Y = Xbf @ W^T   (M=8192, N=1024, K=1024)
//    One wave computes a 32x64 tile: 2 A-frags reuse every B-frag (8 WMMA /
//    k-step). mode 0: bf16 [B,H,T,HD] (Q,K). mode 1: bf16 [B,H,HD,T] (V^T).
// ---------------------------------------------------------------------------
__global__ void qkv_gemm_kernel(const unsigned short* __restrict__ A,
                                const unsigned short* __restrict__ W,
                                unsigned short* __restrict__ Out, int mode) {
    const int lane = threadIdx.x & 31;
    const int wid  = threadIdx.x >> 5;
    const int gid  = blockIdx.x * 4 + wid;          // 4096 wave tiles
    const int m0   = (gid >> 4) * 32;               // 256 M tiles of 32
    const int n0   = (gid & 15) * 64;               // 16 N tiles of 64
    const int nl   = lane & 15;
    const int hih  = lane >> 4;
    const int kbA  = hih ? 16 : 0;
    const int kbB  = hih ? 16 : 0;

    v8f c[8] = {};
    const unsigned short* arow0 = A + (size_t)(m0 + nl) * D_;
    const unsigned short* arow1 = A + (size_t)(m0 + 16 + nl) * D_;

    for (int k0 = 0; k0 < D_; k0 += 32) {
        if (k0 + 32 < D_) {
            __builtin_prefetch(arow0 + k0 + 32, 0, 3);
            __builtin_prefetch(arow1 + k0 + 32, 0, 3);
        }
        v16bf a0 = ld_afrag(arow0 + k0 + kbA);
        v16bf a1 = ld_afrag(arow1 + k0 + kbA);
#pragma unroll
        for (int j = 0; j < 4; ++j) {
            const unsigned short* wp =
                W + (size_t)(n0 + j * 16 + nl) * D_ + k0 + kbB;
            v16bf b = *(const v16bf*)wp;   // row n of W == contiguous K-run
            c[j]     = wmma_bf16(a0, b, c[j]);
            c[4 + j] = wmma_bf16(a1, b, c[4 + j]);
        }
    }

#pragma unroll
    for (int j = 0; j < 4; ++j) {
        const int gn = n0 + j * 16 + nl;
        const int h  = gn >> 6;
        const int hd = gn & 63;
#pragma unroll
        for (int half = 0; half < 2; ++half) {
#pragma unroll
            for (int r = 0; r < 8; ++r) {
                const int gm = m0 + half * 16 + r + 8 * hih;
                const int b  = gm >> 11;
                const int t  = gm & (T_ - 1);
                size_t idx = (mode == 0)
                    ? ((size_t)((b * H_ + h) * T_ + t) * HD_ + hd)
                    : ((size_t)((b * H_ + h) * HD_ + hd) * T_ + t);
                Out[idx] = f2bf(c[half * 4 + j][r]);
            }
        }
    }
}

// ---------------------------------------------------------------------------
// 2) Flash attention: one wave per (b,h, 16-row q tile). kv steps of 32.
//    Double-buffered TDM: while computing on buffer b, the next K tile
//    (32x64 bf16, contiguous 4KB, 1-D D#) and V^T tile (64 rows x 64B,
//    2-D strided D#) stream into buffer b^1. s_wait_tensorcnt 2 gates the
//    current pair (in-order completion); s_wait_dscnt 0 fences LDS reuse.
// ---------------------------------------------------------------------------
struct WaveLds {
    unsigned short kt[2][32 * 64];   // [buf][token][hd]   2 x 4KB
    unsigned short vt[2][64 * 32];   // [buf][hd][token]   2 x 4KB
    unsigned short p [16 * 32];      // P tile             1KB
};

__global__ void attn_kernel(const unsigned short* __restrict__ Q,
                            const unsigned short* __restrict__ K,
                            const unsigned short* __restrict__ V,
                            unsigned short* __restrict__ Ctx) {
    __shared__ __align__(128) WaveLds smem[2];     // 2 waves per block

    const int lane = threadIdx.x & 31;
    const int wid  = threadIdx.x >> 5;
    const int gid  = blockIdx.x * 2 + wid;          // 8192 wave tiles
    const int bh   = gid >> 7;                      // 0..63
    const int q0   = (gid & 127) * 16;
    const int b    = bh >> 4;
    const int h    = bh & 15;
    const int nl   = lane & 15;
    const int hih  = lane >> 4;
    const int kbA  = hih ? 16 : 0;
    const int kbB  = hih ? 16 : 0;

    const unsigned short* Qb = Q + (size_t)bh * T_ * HD_;
    const unsigned short* Kb = K + (size_t)bh * T_ * HD_;
    const unsigned short* Vb = V + (size_t)bh * HD_ * T_;
    WaveLds* L = &smem[wid];

    // Q A-fragments (K-dim = HD = 64 -> two 32-wide k-steps)
    const unsigned short* qrow = Qb + (size_t)(q0 + nl) * HD_;
    const v16bf aQ0 = ld_afrag(qrow + 0  + kbA);
    const v16bf aQ1 = ld_afrag(qrow + 32 + kbA);

#ifdef HAVE_TDM
    const unsigned ldsK[2] = { (unsigned)(unsigned long long)(L->kt[0]),
                               (unsigned)(unsigned long long)(L->kt[1]) };
    const unsigned ldsV[2] = { (unsigned)(unsigned long long)(L->vt[0]),
                               (unsigned)(unsigned long long)(L->vt[1]) };
    const v8i g1K = tdm_g1(/*td0=*/512, /*td1=*/1,  /*tile0=*/512, /*tile1=*/1,  /*stride0=*/512);
    const v8i g1V = tdm_g1(/*td0=*/512, /*td1=*/64, /*tile0=*/8,   /*tile1=*/64, /*stride0=*/512);
    // prologue: stream first K/V tiles into buffer 0
    tdm_load(tdm_g0(ldsK[0], (unsigned long long)(Kb)), g1K);
    tdm_load(tdm_g0(ldsV[0], (unsigned long long)(Vb)), g1V);
#endif

    v8f o[4] = {};
    float mi[8], li[8];
#pragma unroll
    for (int r = 0; r < 8; ++r) { mi[r] = -1e38f; li[r] = 0.0f; }

    const int   qhi   = q0 + 15;
    const float scale = 0.125f;                      // 1/sqrt(64)
    int buf = 0;

    for (int kv0 = 0; kv0 <= qhi; kv0 += 32) {
        const bool hasNext = (kv0 + 32 <= qhi);

        // reads from buf^1 (previous step) and P-tile ops must be complete
        asm volatile("s_wait_dscnt 0x0" ::: "memory");

#ifdef HAVE_TDM
        if (hasNext) {
            tdm_load(tdm_g0(ldsK[buf ^ 1],
                     (unsigned long long)(Kb + (size_t)(kv0 + 32) * HD_)), g1K);
            tdm_load(tdm_g0(ldsV[buf ^ 1],
                     (unsigned long long)(Vb + kv0 + 32)),                 g1V);
            __builtin_amdgcn_s_wait_tensorcnt(2);   // current pair complete
        } else {
            __builtin_amdgcn_s_wait_tensorcnt(0);
        }
        asm volatile("" ::: "memory");
#else
        // fallback: per-lane 8B copies into the current buffer
        {
            const unsigned long long* gk =
                (const unsigned long long*)(Kb + (size_t)kv0 * HD_);
            unsigned long long* lk = (unsigned long long*)L->kt[buf];
            for (int i = lane; i < 512; i += 32) lk[i] = gk[i];
            unsigned long long* lv = (unsigned long long*)L->vt[buf];
            for (int i = lane; i < 256; i += 32) {
                const int row = i >> 2, c8 = i & 3;
                lv[i] = *(const unsigned long long*)(Vb + (size_t)row * T_ +
                                                     kv0 + c8 * 4);
            }
        }
        asm volatile("s_wait_dscnt 0x0" ::: "memory");
#endif

        const unsigned short* kt = L->kt[buf];
        const unsigned short* vt = L->vt[buf];

        // ---- scores via WMMA from LDS K tile ----
        v8f S0 = {};
        {
            const unsigned short* kp = kt + (size_t)nl * HD_ + kbB;
            S0 = wmma_bf16(aQ0, *(const v16bf*)(kp),      S0);
            S0 = wmma_bf16(aQ1, *(const v16bf*)(kp + 32), S0);
        }
        v8f S1 = {};
        const bool v1 = (kv0 + 16 <= qhi);
        if (v1) {
            const unsigned short* kp = kt + (size_t)(16 + nl) * HD_ + kbB;
            S1 = wmma_bf16(aQ0, *(const v16bf*)(kp),      S1);
            S1 = wmma_bf16(aQ1, *(const v16bf*)(kp + 32), S1);
        }

        // ---- online softmax over the 32 new columns ----
#pragma unroll
        for (int r = 0; r < 8; ++r) {
            const int qr = q0 + r + 8 * hih;
            float s0 = S0[r] * scale;
            if (kv0 + nl > qr) s0 = -1e30f;
            float s1 = -1e30f;
            if (v1) {
                s1 = S1[r] * scale;
                if (kv0 + 16 + nl > qr) s1 = -1e30f;
            }
            float rm = fmaxf(s0, s1);
            rm = fmaxf(rm, __shfl_xor(rm, 1, 32));
            rm = fmaxf(rm, __shfl_xor(rm, 2, 32));
            rm = fmaxf(rm, __shfl_xor(rm, 4, 32));
            rm = fmaxf(rm, __shfl_xor(rm, 8, 32));
            const float nm = fmaxf(mi[r], rm);
            const float al = __expf(mi[r] - nm);
            const float p0 = __expf(s0 - nm);
            const float p1 = __expf(s1 - nm);
            float rs = p0 + p1;
            rs += __shfl_xor(rs, 1, 32);
            rs += __shfl_xor(rs, 2, 32);
            rs += __shfl_xor(rs, 4, 32);
            rs += __shfl_xor(rs, 8, 32);
            li[r] = li[r] * al + rs;
            mi[r] = nm;
            o[0][r] *= al; o[1][r] *= al; o[2][r] *= al; o[3][r] *= al;
            const int row = r + 8 * hih;
            L->p[row * 32 + nl]      = f2bf(p0);
            L->p[row * 32 + 16 + nl] = f2bf(p1);
        }

        asm volatile("s_wait_dscnt 0x0" ::: "memory");

        // P as A-fragment (16x32 bf16) from LDS
        const unsigned short* lp = L->p + nl * 32 + kbA;
        v8bf plo = *(const v8bf*)(lp);
        v8bf phi = *(const v8bf*)(lp + 8);
        v16bf aP = cat8(plo, phi);
        asm volatile("" ::: "memory");

        // ---- O += P @ V from LDS V^T tile ----
#pragma unroll
        for (int j = 0; j < 4; ++j) {
            const unsigned short* vp = vt + (size_t)(j * 16 + nl) * 32 + kbB;
            o[j] = wmma_bf16(aP, *(const v16bf*)vp, o[j]);
        }

        buf ^= 1;
    }

    // ---- normalize and store ctx as bf16 [B,T,D] ----
#pragma unroll
    for (int r = 0; r < 8; ++r) {
        const float inv = 1.0f / li[r];
        o[0][r] *= inv; o[1][r] *= inv; o[2][r] *= inv; o[3][r] *= inv;
    }
#pragma unroll
    for (int j = 0; j < 4; ++j) {
#pragma unroll
        for (int r = 0; r < 8; ++r) {
            const int t = q0 + r + 8 * hih;
            size_t idx = ((size_t)b * T_ + t) * D_ + h * HD_ + j * 16 + nl;
            Ctx[idx] = f2bf(o[j][r]);
        }
    }
}

// ---------------------------------------------------------------------------
// 3) Output projection:  out = ctx @ Wo^T + bo   (f32 output, 32x64 tiles)
// ---------------------------------------------------------------------------
__global__ void out_gemm_kernel(const unsigned short* __restrict__ A,
                                const unsigned short* __restrict__ W,
                                const float* __restrict__ bias,
                                float* __restrict__ Out) {
    const int lane = threadIdx.x & 31;
    const int wid  = threadIdx.x >> 5;
    const int gid  = blockIdx.x * 4 + wid;
    const int m0   = (gid >> 4) * 32;
    const int n0   = (gid & 15) * 64;
    const int nl   = lane & 15;
    const int hih  = lane >> 4;
    const int kbA  = hih ? 16 : 0;
    const int kbB  = hih ? 16 : 0;

    v8f c[8] = {};
    const unsigned short* arow0 = A + (size_t)(m0 + nl) * D_;
    const unsigned short* arow1 = A + (size_t)(m0 + 16 + nl) * D_;

    for (int k0 = 0; k0 < D_; k0 += 32) {
        if (k0 + 32 < D_) {
            __builtin_prefetch(arow0 + k0 + 32, 0, 3);
            __builtin_prefetch(arow1 + k0 + 32, 0, 3);
        }
        v16bf a0 = ld_afrag(arow0 + k0 + kbA);
        v16bf a1 = ld_afrag(arow1 + k0 + kbA);
#pragma unroll
        for (int j = 0; j < 4; ++j) {
            const unsigned short* wp =
                W + (size_t)(n0 + j * 16 + nl) * D_ + k0 + kbB;
            v16bf bfrag = *(const v16bf*)wp;
            c[j]     = wmma_bf16(a0, bfrag, c[j]);
            c[4 + j] = wmma_bf16(a1, bfrag, c[4 + j]);
        }
    }

#pragma unroll
    for (int j = 0; j < 4; ++j) {
        const int gn = n0 + j * 16 + nl;
        const float bb = bias[gn];
#pragma unroll
        for (int half = 0; half < 2; ++half) {
#pragma unroll
            for (int r = 0; r < 8; ++r) {
                const int gm = m0 + half * 16 + r + 8 * hih;
                Out[(size_t)gm * D_ + gn] = c[half * 4 + j][r] + bb;
            }
        }
    }
}

// ---------------------------------------------------------------------------
// Launch
// ---------------------------------------------------------------------------
extern "C" void kernel_launch(void* const* d_in, const int* in_sizes, int n_in,
                              void* d_out, int out_size, void* d_ws, size_t ws_size,
                              hipStream_t stream) {
    (void)in_sizes; (void)n_in; (void)out_size; (void)ws_size;

    const float* x  = (const float*)d_in[0];
    const float* Wq = (const float*)d_in[1];
    const float* Wk = (const float*)d_in[2];
    const float* Wv = (const float*)d_in[3];
    const float* Wo = (const float*)d_in[4];
    const float* bo = (const float*)d_in[5];
    float* out = (float*)d_out;

    char* ws = (char*)d_ws;
    const size_t xN = (size_t)M_ * D_;
    const size_t wN = (size_t)D_ * D_;

    unsigned short* xb  = (unsigned short*)(ws);                 ws += xN * 2;
    unsigned short* wqb = (unsigned short*)(ws);                 ws += wN * 2;
    unsigned short* wkb = (unsigned short*)(ws);                 ws += wN * 2;
    unsigned short* wvb = (unsigned short*)(ws);                 ws += wN * 2;
    unsigned short* wob = (unsigned short*)(ws);                 ws += wN * 2;
    unsigned short* Qb  = (unsigned short*)(ws);                 ws += xN * 2;
    unsigned short* Kb  = (unsigned short*)(ws);                 ws += xN * 2;
    unsigned short* Vb  = (unsigned short*)(ws);                 ws += xN * 2;
    unsigned short* Cx  = (unsigned short*)(ws);                 ws += xN * 2;

    cvt_bf16_kernel<<<(int)((xN + 255) / 256), 256, 0, stream>>>(x,  xb,  (int)xN);
    cvt_bf16_kernel<<<(int)((wN + 255) / 256), 256, 0, stream>>>(Wq, wqb, (int)wN);
    cvt_bf16_kernel<<<(int)((wN + 255) / 256), 256, 0, stream>>>(Wk, wkb, (int)wN);
    cvt_bf16_kernel<<<(int)((wN + 255) / 256), 256, 0, stream>>>(Wv, wvb, (int)wN);
    cvt_bf16_kernel<<<(int)((wN + 255) / 256), 256, 0, stream>>>(Wo, wob, (int)wN);

    // 4096 wave tiles (32x64) -> 1024 blocks x 4 waves
    qkv_gemm_kernel<<<1024, 128, 0, stream>>>(xb, wqb, Qb, 0);
    qkv_gemm_kernel<<<1024, 128, 0, stream>>>(xb, wkb, Kb, 0);
    qkv_gemm_kernel<<<1024, 128, 0, stream>>>(xb, wvb, Vb, 1);

    // 64 bh * 128 q-tiles = 8192 wave tiles -> 4096 blocks x 2 waves
    attn_kernel<<<4096, 64, 0, stream>>>(Qb, Kb, Vb, Cx);

    out_gemm_kernel<<<1024, 128, 0, stream>>>(Cx, wob, bo, out);
}
